// GCNConv_21818433863981
// MI455X (gfx1250) — matrix operand order repaired and checked by reading, not compiled, not run on hardware
//
#include <hip/hip_runtime.h>
#include <hip/hip_bf16.h>

// ---------------------------------------------------------------------------
// GCNConv forward for MI455X (gfx1250, wave32, WMMA).
//   h   = x @ W            (bf16 WMMA, f32 accumulate)   -> d_ws
//   out = segment_sum(vals * h[cols], rows) + bias       (L2 f32 atomics)
// ---------------------------------------------------------------------------

typedef __attribute__((ext_vector_type(16))) __bf16 v16bf;
typedef __attribute__((ext_vector_type(8)))  __bf16 v8bf;
typedef __attribute__((ext_vector_type(8)))  float  v8f;

#define N_NODES 100000
#define N_EDGES 1600000
#define IN_F    256
#define OUT_F   128

#define GEMM_BLOCKS 1250           // 5 M-tiles of 16 rows per block: 1250*5*16 = 100000
#define M_ITERS     5
#define WT_STRIDE   264            // 256 + 8 bf16 pad -> conflict-free column reads
#define SMEM_BYTES  ((128 * WT_STRIDE + 16 * WT_STRIDE) * 2)   // Wt + Xs, bf16

// --------------------------------------------------------------------------
// Kernel 0: out[i] = bias[i % 128]   (segment-sum base; d_out is poisoned)
// --------------------------------------------------------------------------
__global__ __launch_bounds__(256) void gcn_init_kernel(
    const float* __restrict__ bias, float* __restrict__ out) {
  size_t i = (size_t)blockIdx.x * 256 + threadIdx.x;   // grid sized exactly
  out[i] = bias[i & (OUT_F - 1)];
}

// --------------------------------------------------------------------------
// Kernel 1: h = x @ W via v_wmma_f32_16x16x32_bf16.
//  - Weight staged transposed in LDS as bf16 (Wt[n][k], stride 264).
//  - x tile (16 rows x 256) staged in LDS as bf16 per M-tile.
//  - 8 waves/block; wave w computes the 16x16 tile at columns [16w, 16w+16).
// --------------------------------------------------------------------------
__global__ __launch_bounds__(256) void gcn_gemm_kernel(
    const float* __restrict__ x, const float* __restrict__ w,
    float* __restrict__ h) {
  extern __shared__ __bf16 smem[];
  __bf16* Wt = smem;                        // [128][WT_STRIDE]
  __bf16* Xs = smem + 128 * WT_STRIDE;      // [ 16][WT_STRIDE]

  const int tid  = threadIdx.x;
  const int lane = tid & 31;
  const int wv   = tid >> 5;                // n-tile index, 0..7

  // Stage full weight, transposed, as bf16 (once per block).
  for (int i = tid; i < IN_F * OUT_F; i += 256) {
    int k = i >> 7;                         // i / 128
    int n = i & 127;
    Wt[n * WT_STRIDE + k] = (__bf16)w[i];
  }

  // Per-lane fragment geometry (A/B in contiguous K order: k = 16*(lane/16) + s).
  const int m_l   = lane & 15;              // A row within tile
  const int hiA   = (lane >> 4) << 3;       // 0 or 8  (C row-half offset)
  const int n_l   = lane & 15;              // B/C column within tile
  const int koffB = (lane >> 4) << 4;       // 0 or 16 (B half-K select)

  for (int it = 0; it < M_ITERS; ++it) {
    const int m_base = (blockIdx.x + it * GEMM_BLOCKS) << 4;

    __syncthreads();                        // weight ready / prior tile reads done
    {
      // Stage x[m_base .. m_base+15][0..255] as bf16: thread t loads a 16-elem chunk.
      const int row = tid >> 4, ch = tid & 15;
      const float* src = x + (size_t)(m_base + row) * IN_F + ch * 16;
      __bf16* dst = Xs + row * WT_STRIDE + ch * 16;
#pragma unroll
      for (int j = 0; j < 16; ++j) dst[j] = (__bf16)src[j];
    }
    __syncthreads();

    v8f c = {};                             // f32 accumulator, 16x16 tile
#pragma unroll
    for (int kb = 0; kb < IN_F / 32; ++kb) {
      // A fragment: row m_l, 16 contiguous K values [koffB, koffB+16). 16B aligned.
      const __bf16* arow = Xs + m_l * WT_STRIDE + kb * 32;
      v8bf a0 = *(const v8bf*)(arow + koffB);
      v8bf a1 = *(const v8bf*)(arow + koffB + 8);
      v16bf a = __builtin_shufflevector(a0, a1, 0, 1, 2, 3, 4, 5, 6, 7,
                                               8, 9, 10, 11, 12, 13, 14, 15);
      // B fragment: column (16*wv + n_l), 16 contiguous K values. 16B aligned.
      const __bf16* brow = Wt + (wv * 16 + n_l) * WT_STRIDE + kb * 32 + koffB;
      v8bf b0 = *(const v8bf*)(brow);
      v8bf b1 = *(const v8bf*)(brow + 8);
      v16bf b = __builtin_shufflevector(b0, b1, 0, 1, 2, 3, 4, 5, 6, 7,
                                               8, 9, 10, 11, 12, 13, 14, 15);
      c = __builtin_amdgcn_wmma_f32_16x16x32_bf16(
          /*neg_a=*/false, a, /*neg_b=*/false, b,
          /*c_mod=*/(short)0, c, /*reuse_a=*/false, /*reuse_b=*/false);
    }

    // C layout: VGPR r holds row (r + hiA), column n_l.
    float* hout = h + (size_t)m_base * OUT_F + wv * 16;
#pragma unroll
    for (int r = 0; r < 8; ++r)
      hout[(size_t)(r + hiA) * OUT_F + n_l] = c[r];
  }
}

// --------------------------------------------------------------------------
// Kernel 2: one wave32 per edge. Lane l owns features [4l, 4l+4).
//   out[rows[e]] += vals[e] * h[cols[e]]   (relaxed agent-scope f32 atomics,
//   both h and out resident in the 192MB L2 -> L2-bandwidth bound)
// --------------------------------------------------------------------------
__global__ __launch_bounds__(256) void gcn_scatter_kernel(
    const float* __restrict__ h, const int* __restrict__ rows,
    const int* __restrict__ cols, const float* __restrict__ vals,
    float* __restrict__ out, int n_edges) {
  const int lane = threadIdx.x & 31;
  const int e = blockIdx.x * (256 / 32) + (threadIdx.x >> 5);
  if (e >= n_edges) return;                 // uniform per wave

  const int   r = rows[e];
  const int   c = cols[e];
  const float v = vals[e];

  const float4* hrow = (const float4*)(h + (size_t)c * OUT_F);
  float4 m = hrow[lane];

  float* o = out + (size_t)r * OUT_F + lane * 4;
  __hip_atomic_fetch_add(o + 0, m.x * v, __ATOMIC_RELAXED, __HIP_MEMORY_SCOPE_AGENT);
  __hip_atomic_fetch_add(o + 1, m.y * v, __ATOMIC_RELAXED, __HIP_MEMORY_SCOPE_AGENT);
  __hip_atomic_fetch_add(o + 2, m.z * v, __ATOMIC_RELAXED, __HIP_MEMORY_SCOPE_AGENT);
  __hip_atomic_fetch_add(o + 3, m.w * v, __ATOMIC_RELAXED, __HIP_MEMORY_SCOPE_AGENT);
}

// --------------------------------------------------------------------------
extern "C" void kernel_launch(void* const* d_in, const int* in_sizes, int n_in,
                              void* d_out, int out_size, void* d_ws, size_t ws_size,
                              hipStream_t stream) {
  const float* x         = (const float*)d_in[0];
  const int*   edge_rows = (const int*)d_in[1];
  const int*   edge_cols = (const int*)d_in[2];
  const float* edge_vals = (const float*)d_in[3];
  const float* weight    = (const float*)d_in[4];
  const float* bias      = (const float*)d_in[5];
  float*       out       = (float*)d_out;
  float*       h         = (float*)d_ws;    // needs N_NODES*OUT_F*4 = 51.2 MB

  // 1) out = bias (broadcast).  12.8M elements / 256 = 50000 blocks exactly.
  gcn_init_kernel<<<(N_NODES * OUT_F) / 256, 256, 0, stream>>>(bias, out);

  // 2) h = x @ W  (bf16 WMMA, f32 accumulate).
  gcn_gemm_kernel<<<GEMM_BLOCKS, 256, SMEM_BYTES, stream>>>(x, weight, h);

  // 3) scatter-add edges.  8 edges per 256-thread block -> 200000 blocks.
  gcn_scatter_kernel<<<(N_EDGES + 7) / 8, 256, 0, stream>>>(
      h, edge_rows, edge_cols, edge_vals, out, N_EDGES);
}
